// ExpertChoiceMoEMatcher_60017872994619
// MI455X (gfx1250) — compile-verified, matrix operand order checked by reading, not run on hardware
//
#include <hip/hip_runtime.h>

typedef __attribute__((ext_vector_type(16))) _Float16 v16h;
typedef __attribute__((ext_vector_type(8)))  float    v8f;

#define B_T   131072
#define DDIM  256
#define NEXP  16
#define KSEL  8192           // B_T / NEXP
#define NBIN  2048           // 11-bit radix histogram

// ---- d_out layout (floats, concatenated tuple outputs) ----
#define OUT_ACT 0                          // [B_T, D] complex64 -> interleaved (r,i) floats
#define OUT_IDX ((size_t)B_T * DDIM * 2)   // [KSEL, E] indices (as float)
#define OUT_SC  (OUT_IDX + (size_t)KSEL * NEXP)
#define OUT_CNT (OUT_SC  + (size_t)KSEL * NEXP)

// ---- ws layout (bytes) ----
#define WS_SCORES 0u                       // B_T*E f32      = 8388608 B
#define WS_HIST   8388608u                 // E*2048 u32     = 131072 B
#define WS_THR    8519680u                 // E i32
#define WS_QUOTA  8519744u                 // E i32
#define WS_IDX    8519808u                 // KSEL*E i32     = 524288 B
#define WS_GW     9044096u                 // 16*32*16 f16   = 16384 B
#define WS_WR     9060480u                 // E*16*8*32*16 f16 = 2097152 B
#define WS_WI     11157632u                // same

static __device__ inline v8f wmma_f16(v16h a, v16h b, v8f c) {
    return __builtin_amdgcn_wmma_f32_16x16x32_f16(false, a, false, b, (short)0, c,
                                                  false, false);
}

// A-fragment (16-bit, 16x32) per-lane element j -> K index:
//   j<8:  k = hi*8 + j        j>=8: k = 16 + hi*8 + (j-8)
static __device__ inline v16h load_a_frag(const float* row /* k-tile base of this lane's M row */,
                                          int hi) {
    __align__(16) float tmp[16];
    *(float4*)&tmp[0]  = *(const float4*)(row + hi * 8);
    *(float4*)&tmp[4]  = *(const float4*)(row + hi * 8 + 4);
    *(float4*)&tmp[8]  = *(const float4*)(row + 16 + hi * 8);
    *(float4*)&tmp[12] = *(const float4*)(row + 16 + hi * 8 + 4);
    v16h a;
#pragma unroll
    for (int j = 0; j < 16; ++j) a[j] = (_Float16)tmp[j];
    return a;
}

static __device__ inline unsigned score_key(float f) {
    unsigned u = __float_as_uint(f);
    return (u & 0x80000000u) ? ~u : (u | 0x80000000u);  // ascending key == ascending float
}

// ---------------- zero fill ----------------
__global__ void k_zero_f(float* __restrict__ p, long long n) {
    long long i = (long long)blockIdx.x * blockDim.x + threadIdx.x;
    long long stride = (long long)gridDim.x * blockDim.x;
    for (; i < n; i += stride) p[i] = 0.0f;
}

// ---------------- pack gate_w [512][16] f32 -> fragment layout [kt(16)][lane(32)][j(16)] f16 ----
__global__ void k_pack_gate(const float* __restrict__ gw, _Float16* __restrict__ gwp) {
    int g = blockIdx.x * blockDim.x + threadIdx.x;       // 512 entries
    if (g >= 16 * 32) return;
    int kt = g >> 5, lane = g & 31, hi = lane >> 4, n = lane & 15;
#pragma unroll
    for (int j = 0; j < 16; ++j) {
        int k = kt * 32 + ((j < 8) ? (hi * 8 + j) : (16 + hi * 8 + (j - 8)));
        gwp[(size_t)g * 16 + j] = (_Float16)gw[k * NEXP + n];
    }
}

// ---------------- pack experts_w [E,D,D,2] f32 -> wr/wi [e][nt(16)][kt(8)][lane(32)][j(16)] f16 ----
__global__ void k_pack_w(const float* __restrict__ w, _Float16* __restrict__ wr,
                         _Float16* __restrict__ wi) {
    int g = blockIdx.x * blockDim.x + threadIdx.x;       // E*16*8*32 = 65536 entries
    if (g >= NEXP * 16 * 8 * 32) return;
    int lane = g & 31, kt = (g >> 5) & 7, nt = (g >> 8) & 15, e = g >> 12;
    int hi = lane >> 4, n16 = lane & 15;
#pragma unroll
    for (int j = 0; j < 16; ++j) {
        int k = kt * 32 + ((j < 8) ? (hi * 8 + j) : (16 + hi * 8 + (j - 8)));
        int n = nt * 16 + n16;
        const float* src = w + (((size_t)e * DDIM + k) * DDIM + n) * 2;
        wr[(size_t)g * 16 + j] = (_Float16)src[0];
        wi[(size_t)g * 16 + j] = (_Float16)src[1];
    }
}

// ---------------- gating GEMM: scores[B_T][16] = [xr|xi] @ gate_w, WMMA f16 ----------------
__global__ __launch_bounds__(256) void k_gate(const float* __restrict__ xr,
                                              const float* __restrict__ xi,
                                              const _Float16* __restrict__ gwp,
                                              float* __restrict__ scores) {
    int wave = threadIdx.x >> 5, lane = threadIdx.x & 31;
    int hi = lane >> 4, n16 = lane & 15;
    int tile = blockIdx.x * 8 + wave;                    // 16 tokens per wave
    int trow = tile * 16 + n16;                          // A row (M = lane&15)
    v8f acc = {};
#pragma unroll
    for (int kt = 0; kt < 16; ++kt) {
        int k0 = kt * 32;
        const float* row = (k0 < DDIM) ? (xr + (size_t)trow * DDIM + k0)
                                       : (xi + (size_t)trow * DDIM + (k0 - DDIM));
        v16h a = load_a_frag(row, hi);
        v16h b = *(const v16h*)(gwp + ((size_t)kt * 32 + lane) * 16);
        acc = wmma_f16(a, b, acc);
    }
#pragma unroll
    for (int v = 0; v < 8; ++v) {
        int m = v + 8 * hi;                              // C row in this lane's VGPR v
        scores[(size_t)(tile * 16 + m) * NEXP + n16] = acc[v];
    }
}

// ---------------- per-expert 2048-bin histogram of score keys ----------------
__global__ void k_hist(const float* __restrict__ scores, unsigned* __restrict__ hist) {
    long long n = (long long)B_T * NEXP;
    long long i = (long long)blockIdx.x * blockDim.x + threadIdx.x;
    long long stride = (long long)gridDim.x * blockDim.x;
    for (; i < n; i += stride) {
        int e = (int)(i & (NEXP - 1));
        unsigned bin = score_key(scores[i]) >> 21;
        atomicAdd(&hist[e * NBIN + bin], 1u);
    }
}

// ---------------- find threshold bin + tie quota per expert ----------------
__global__ void k_thresh(const unsigned* __restrict__ hist, int* __restrict__ thr,
                         int* __restrict__ quota) {
    int e = threadIdx.x;
    if (e >= NEXP) return;
    const unsigned* h = hist + e * NBIN;
    unsigned cum = 0;
    int b = NBIN - 1;
    for (; b > 0; --b) {
        cum += h[b];
        if (cum >= (unsigned)KSEL) break;
    }
    if (cum < (unsigned)KSEL) cum += h[0];               // b == 0 fallthrough
    thr[e] = b;
    quota[e] = KSEL - (int)(cum - h[b]);
}

// ---------------- ordered compaction: one wave per expert (deterministic) ----------------
__global__ void k_select(const float* __restrict__ scores, const int* __restrict__ thr,
                         const int* __restrict__ quota, int* __restrict__ idxws,
                         float* __restrict__ out_idx, float* __restrict__ out_sc,
                         float* __restrict__ counts) {
    int e = blockIdx.x;
    int lane = threadIdx.x;
    int tb = thr[e];
    unsigned q = (unsigned)quota[e];
    unsigned lt = (1u << lane) - 1u;
    unsigned base = 0, eqtaken = 0;
    for (int chunk = 0; chunk < B_T; chunk += 32) {
        int t = chunk + lane;
        float f = scores[(size_t)t * NEXP + e];
        int bin = (int)(score_key(f) >> 21);
        bool gt = bin > tb;
        bool eq = bin == tb;
        unsigned meq = __builtin_amdgcn_ballot_w32(eq);
        unsigned rank_eq = (unsigned)__popc(meq & lt);
        bool take_eq = eq && (eqtaken + rank_eq < q);
        bool sel = gt || take_eq;
        unsigned msel = __builtin_amdgcn_ballot_w32(sel);
        unsigned pos = base + (unsigned)__popc(msel & lt);
        if (sel && pos < (unsigned)KSEL) {
            idxws[pos * NEXP + e] = t;
            out_idx[pos * NEXP + e] = (float)t;
            out_sc[pos * NEXP + e] = f;
            atomicAdd(&counts[t], 1.0f);                 // exact: sum of 1.0s
        }
        base += (unsigned)__popc(msel);
        eqtaken += (unsigned)__popc(__builtin_amdgcn_ballot_w32(take_eq));
    }
}

// ---------------- expert complex GEMM + weighted scatter-add (WMMA f16) ----------------
__global__ __launch_bounds__(256) void k_expert_gemm(
    const float* __restrict__ xr, const float* __restrict__ xi,
    const int* __restrict__ idxws, const float* __restrict__ out_sc,
    const _Float16* __restrict__ wr, const _Float16* __restrict__ wi,
    float* __restrict__ out) {
    __shared__ __align__(32) _Float16 s_ar[8][32][16];   // 8 KB
    __shared__ __align__(32) _Float16 s_ai[8][32][16];   // 8 KB
    __shared__ int   s_tok[16];
    __shared__ float s_sc[16];

    int e = blockIdx.x >> 9;                             // 512 token-tiles per expert
    int tile = blockIdx.x & 511;
    int tid = threadIdx.x;

    if (tid < 16) {
        int pos = tile * 16 + tid;
        s_tok[tid] = idxws[pos * NEXP + e];
        s_sc[tid] = out_sc[pos * NEXP + e];
    }
    __syncthreads();

    {   // stage gathered A tile (16 tokens x 256 K, r & i) into fragment-swizzled LDS
        int kt = tid >> 5, lane = tid & 31, m = lane & 15, hi = lane >> 4;
        size_t t = (size_t)s_tok[m];
        const float* rr = xr + t * DDIM + kt * 32;
        const float* ri = xi + t * DDIM + kt * 32;
        __align__(16) float tr[16], ti[16];
        *(float4*)&tr[0]  = *(const float4*)(rr + hi * 8);
        *(float4*)&tr[4]  = *(const float4*)(rr + hi * 8 + 4);
        *(float4*)&tr[8]  = *(const float4*)(rr + 16 + hi * 8);
        *(float4*)&tr[12] = *(const float4*)(rr + 16 + hi * 8 + 4);
        *(float4*)&ti[0]  = *(const float4*)(ri + hi * 8);
        *(float4*)&ti[4]  = *(const float4*)(ri + hi * 8 + 4);
        *(float4*)&ti[8]  = *(const float4*)(ri + 16 + hi * 8);
        *(float4*)&ti[12] = *(const float4*)(ri + 16 + hi * 8 + 4);
#pragma unroll
        for (int j = 0; j < 16; ++j) {
            s_ar[kt][lane][j] = (_Float16)tr[j];
            s_ai[kt][lane][j] = (_Float16)ti[j];
        }
    }
    __syncthreads();

    int wave = tid >> 5, lane = tid & 31, hi = lane >> 4, n16 = lane & 15;
#pragma unroll
    for (int half = 0; half < 2; ++half) {
        int nt = wave + half * 8;                        // this wave's N tile (2 of 16)
        const v16h* br = (const v16h*)wr + ((size_t)e * 16 + nt) * 8 * 32;
        const v16h* bi = (const v16h*)wi + ((size_t)e * 16 + nt) * 8 * 32;
        v8f c1 = {}, c2 = {}, c3 = {};                   // ar*br, ai*bi, ar*bi+ai*br
#pragma unroll
        for (int kt = 0; kt < 8; ++kt) {
            v16h ar = *(const v16h*)&s_ar[kt][lane][0];
            v16h ai = *(const v16h*)&s_ai[kt][lane][0];
            v16h vbr = br[kt * 32 + lane];
            v16h vbi = bi[kt * 32 + lane];
            c1 = wmma_f16(ar, vbr, c1);
            c2 = wmma_f16(ai, vbi, c2);
            c3 = wmma_f16(ar, vbi, c3);
            c3 = wmma_f16(ai, vbr, c3);
        }
#pragma unroll
        for (int v = 0; v < 8; ++v) {
            int m = v + 8 * hi;
            float s = s_sc[m];
            float yr = (c1[v] - c2[v]) * s;
            float yi = c3[v] * s;
            size_t t = (size_t)s_tok[m];
            size_t a = (t * DDIM + (size_t)nt * 16 + n16) * 2;
            atomicAdd(&out[a], yr);
            atomicAdd(&out[a + 1], yi);
        }
    }
}

// ---------------- divide by counts + ModReLU, in place on d_out ----------------
__global__ void k_finalize(float* __restrict__ out, const float* __restrict__ counts,
                           const float* __restrict__ bias) {
    long long n = (long long)B_T * DDIM;
    long long i = (long long)blockIdx.x * blockDim.x + threadIdx.x;
    long long stride = (long long)gridDim.x * blockDim.x;
    for (; i < n; i += stride) {
        long long t = i >> 8;
        int dcol = (int)(i & (DDIM - 1));
        float zr = out[2 * i], zi = out[2 * i + 1];
        float denom = fmaxf(counts[t], 1.0f);
        zr /= denom;
        zi /= denom;
        float mag = sqrtf(zr * zr + zi * zi);
        float safe = fmaxf(mag, 1e-8f);
        float scale = fmaxf(mag + bias[dcol], 0.0f) / safe;
        out[2 * i] = zr * scale;
        out[2 * i + 1] = zi * scale;
    }
}

extern "C" void kernel_launch(void* const* d_in, const int* in_sizes, int n_in,
                              void* d_out, int out_size, void* d_ws, size_t ws_size,
                              hipStream_t stream) {
    const float* xr = (const float*)d_in[0];             // [B_T, D]
    const float* xi = (const float*)d_in[1];             // [B_T, D]
    const float* gw = (const float*)d_in[2];             // [2D, E]
    const float* ew = (const float*)d_in[3];             // [E, D, D, 2]
    const float* mb = (const float*)d_in[4];             // [D]
    float* out = (float*)d_out;

    char* ws = (char*)d_ws;
    float*     scores = (float*)(ws + WS_SCORES);
    unsigned*  hist   = (unsigned*)(ws + WS_HIST);
    int*       thr    = (int*)(ws + WS_THR);
    int*       quota  = (int*)(ws + WS_QUOTA);
    int*       idxws  = (int*)(ws + WS_IDX);
    _Float16*  gwp    = (_Float16*)(ws + WS_GW);
    _Float16*  wr     = (_Float16*)(ws + WS_WR);
    _Float16*  wi     = (_Float16*)(ws + WS_WI);

    // zero the accumulation targets (harness poisons them)
    k_zero_f<<<8192, 256, 0, stream>>>(out + OUT_ACT, (long long)B_T * DDIM * 2);
    k_zero_f<<<512, 256, 0, stream>>>(out + OUT_CNT, (long long)B_T);
    k_zero_f<<<32, 256, 0, stream>>>((float*)hist, (long long)NEXP * NBIN);

    // pack weights to f16 WMMA-fragment layout
    k_pack_gate<<<2, 256, 0, stream>>>(gw, gwp);
    k_pack_w<<<256, 256, 0, stream>>>(ew, wr, wi);

    // gating GEMM (WMMA f16)
    k_gate<<<B_T / 128, 256, 0, stream>>>(xr, xi, gwp, scores);

    // exact per-expert top-k via radix histogram + ordered compaction
    k_hist<<<4096, 256, 0, stream>>>(scores, hist);
    k_thresh<<<1, 32, 0, stream>>>(hist, thr, quota);
    k_select<<<NEXP, 32, 0, stream>>>(scores, thr, quota, idxws, out + OUT_IDX,
                                      out + OUT_SC, out + OUT_CNT);

    // expert complex GEMMs (WMMA f16) with score-weighted atomic scatter-add
    k_expert_gemm<<<NEXP * 512, 256, 0, stream>>>(xr, xi, idxws, out + OUT_SC, wr, wi,
                                                  out + OUT_ACT);

    // combine normalize + ModReLU
    k_finalize<<<4096, 256, 0, stream>>>(out + OUT_ACT, out + OUT_CNT, mb);
}
